// WebAttention_66288525247258
// MI455X (gfx1250) — matrix operand order repaired and checked by reading, not compiled, not run on hardware
//
#include <hip/hip_runtime.h>
#include <math.h>

typedef _Float16 half_t;
typedef __attribute__((ext_vector_type(16))) _Float16 v16h;
typedef __attribute__((ext_vector_type(8)))  _Float16 v8h;
typedef __attribute__((ext_vector_type(8)))  float    v8f;
typedef __attribute__((ext_vector_type(4)))  unsigned int u32x4;
typedef __attribute__((ext_vector_type(8)))  unsigned int u32x8;

#define S_LEN 2048
#define HIDN  2048
#define NH    32
#define NKV   8
#define HD_   64

union V16U { v16h v; v8h h[2]; };

__device__ __forceinline__ v16h make_v16(v8h lo, v8h hi) {
  V16U u; u.h[0] = lo; u.h[1] = hi; return u.v;
}

// A-fragment (16 x K=32 f16) from row-major buffer, leading dim `ld` halves.
// CDNA5 layout: lane m(0-15): halves 0..7 -> K=sel*8+j, halves 8..15 -> K=16+sel*8+j
__device__ __forceinline__ v16h load_frag_a(const half_t* base, int ld, int lane) {
  int m = lane & 15, sel = lane >> 4;
  const half_t* p = base + (size_t)m * ld + sel * 8;
  return make_v16(*(const v8h*)p, *(const v8h*)(p + 16));
}

// B-fragment (K=32 x 16 f16) from buffer laid out element(k,n) = base[n*ld + k]
__device__ __forceinline__ v16h load_frag_b(const half_t* base, int ld, int lane) {
  int n = lane & 15, sel = lane >> 4;
  const half_t* p = base + (size_t)n * ld + sel * 8;
  return make_v16(*(const v8h*)p, *(const v8h*)(p + 16));
}

__device__ __forceinline__ v8f wmma_f16(v16h a, v16h b, v8f c) {
  return __builtin_amdgcn_wmma_f32_16x16x32_f16(false, a, false, b, (short)0, c, false, false);
}

// ---- Tensor Data Mover: DMA a rows x cols (f16) tile, row stride = `stride` halves,
// from global memory into LDS at byte offset lds_off. Issued by one wave; EXEC ignored.
__device__ __forceinline__ void tdm_load_tile_f16(unsigned lds_off, const half_t* gsrc,
                                                  unsigned rows, unsigned cols, unsigned stride) {
#if defined(__HIP_DEVICE_COMPILE__)
  unsigned long long ga = (unsigned long long)(uintptr_t)gsrc;
  u32x4 g0;
  // word0: count=1, is_restore=0, gather off
  g0[0] = (unsigned)__builtin_amdgcn_readfirstlane(1);
  // word1: lds_addr
  g0[1] = (unsigned)__builtin_amdgcn_readfirstlane((int)lds_off);
  // word2/3: global_addr[56:0], type=2 in bits 127:126
  g0[2] = (unsigned)__builtin_amdgcn_readfirstlane((int)(unsigned)ga);
  g0[3] = (unsigned)__builtin_amdgcn_readfirstlane(
      (int)((((unsigned)(ga >> 32)) & 0x01FFFFFFu) | (2u << 30)));
  u32x8 g1;
  g1[0] = (unsigned)__builtin_amdgcn_readfirstlane(1 << 16);  // data_size=1 (2B), mask=0
  g1[1] = (unsigned)__builtin_amdgcn_readfirstlane((int)((cols & 0xFFFFu) << 16));   // tensor_dim0 lo
  g1[2] = (unsigned)__builtin_amdgcn_readfirstlane((int)((cols >> 16) | ((rows & 0xFFFFu) << 16))); // dim0 hi | dim1 lo
  g1[3] = (unsigned)__builtin_amdgcn_readfirstlane((int)((rows >> 16) | ((cols & 0xFFFFu) << 16))); // dim1 hi | tile_dim0
  g1[4] = (unsigned)__builtin_amdgcn_readfirstlane((int)(rows & 0xFFFFu));           // tile_dim1, tile_dim2=0
  g1[5] = (unsigned)__builtin_amdgcn_readfirstlane((int)stride);                     // tensor_dim0_stride lo
  g1[6] = (unsigned)__builtin_amdgcn_readfirstlane(0);
  g1[7] = (unsigned)__builtin_amdgcn_readfirstlane(0);
  asm volatile("tensor_load_to_lds %0, %1" :: "s"(g0), "s"(g1) : "memory");
#else
  (void)lds_off; (void)gsrc; (void)rows; (void)cols; (void)stride;
#endif
}

__device__ __forceinline__ void wait_tensorcnt0() {
#if defined(__HIP_DEVICE_COMPILE__)
  __builtin_amdgcn_s_wait_tensorcnt(0);
#endif
}

__device__ __forceinline__ void prefetch_ro(const void* p) {
#if defined(__HIP_DEVICE_COMPILE__)
  __builtin_prefetch(p, 0, 0);
#else
  (void)p;
#endif
}

// ---------------------------------------------------------------- convert
__global__ void k_f32_to_f16(const float* __restrict__ src, half_t* __restrict__ dst, int n) {
  for (int i = blockIdx.x * blockDim.x + threadIdx.x; i < n; i += gridDim.x * blockDim.x)
    dst[i] = (half_t)src[i];
}

// ---------------------------------------------------------------- GEMM: C[M,N] f32 = A[M,K]f16 * B[K,N]f16
// block = 128 threads (4 waves), tile 64x64, BK=64 -> 8 WMMA per barrier round
__global__ __launch_bounds__(128) void k_gemm(const half_t* __restrict__ A,
                                              const half_t* __restrict__ B,
                                              float* __restrict__ C,
                                              int M, int N, int K) {
  __shared__ __align__(16) half_t sA[64 * 64];   // row-major [row][k]
  __shared__ __align__(16) half_t sBt[64 * 64];  // transposed [n][k]
  int tid = threadIdx.x, lane = tid & 31, wave = tid >> 5;
  int bm = blockIdx.y * 64, bn = blockIdx.x * 64;
  v8f acc[4];
  for (int t = 0; t < 4; ++t)
    for (int r = 0; r < 8; ++r) acc[t][r] = 0.f;

  for (int k0 = 0; k0 < K; k0 += 64) {
    int r = tid >> 1, c = (tid & 1) * 32;
    // stage A: 64x64 halves, 32 halves/thread
    const half_t* ag = A + (size_t)(bm + r) * K + k0 + c;
#pragma unroll
    for (int q = 0; q < 4; ++q)
      *(v8h*)(sA + r * 64 + c + q * 8) = *(const v8h*)(ag + q * 8);
    // stage B transposed: [k][n] in, [n][k] out
    const half_t* bg = B + (size_t)(k0 + r) * N + bn + c;
#pragma unroll
    for (int q = 0; q < 4; ++q) {
      v8h bv = *(const v8h*)(bg + q * 8);
#pragma unroll
      for (int j = 0; j < 8; ++j) sBt[(c + q * 8 + j) * 64 + r] = bv[j];
    }
    __syncthreads();

#pragma unroll
    for (int ks = 0; ks < 64; ks += 32) {
      v16h a = load_frag_a(sA + wave * 16 * 64 + ks, 64, lane);
#pragma unroll
      for (int nt = 0; nt < 4; ++nt) {
        v16h b = load_frag_b(sBt + nt * 16 * 64 + ks, 64, lane);
        acc[nt] = wmma_f16(a, b, acc[nt]);
      }
    }
    __syncthreads();
  }

  int n = lane & 15, mo = (lane >> 4) * 8;
  int m0 = bm + wave * 16;
#pragma unroll
  for (int nt = 0; nt < 4; ++nt)
#pragma unroll
    for (int r = 0; r < 8; ++r)
      C[(size_t)(m0 + mo + r) * N + bn + nt * 16 + n] = acc[nt][r];
}

// ---------------------------------------------------------------- RoPE + RMSNorm -> head-major f16
__global__ __launch_bounds__(64) void k_rope_rms(const float* __restrict__ src,
                                                 half_t* __restrict__ dst,
                                                 const float* __restrict__ lnw, int H) {
  int s = blockIdx.x, h = blockIdx.y, d = threadIdx.x;
  __shared__ float red[64];
  const float* row = src + (size_t)s * H * HD_ + h * HD_;
  float x  = row[d];
  float xp = (d < 32) ? -row[d + 32] : row[d - 32];
  float inv = __powf(10000.f, -(float)(d & 31) * (1.f / 32.f));
  float ang = (float)s * inv;
  float y = x * __cosf(ang) + xp * __sinf(ang);
  red[d] = y * y;
  __syncthreads();
  for (int o = 32; o > 0; o >>= 1) {
    if (d < o) red[d] += red[d + o];
    __syncthreads();
  }
  float var = red[0] * (1.f / 64.f);
  float out = y * rsqrtf(var + 1e-6f) * lnw[d];
  dst[((size_t)h * S_LEN + s) * HD_ + d] = (half_t)out;
}

// ---------------------------------------------------------------- V: [s, kv*64+d] f32 -> [kv][s][d] f16
__global__ void k_v_pack(const float* __restrict__ src, half_t* __restrict__ dst) {
  int i = blockIdx.x * blockDim.x + threadIdx.x;
  int s = i / (NKV * HD_);
  int c = i % (NKV * HD_);
  int h = c / HD_, d = c % HD_;
  dst[((size_t)h * S_LEN + s) * HD_ + d] = (half_t)src[i];
}

// ---------------------------------------------------------------- Flash attention
// grid (S/64, NH), block 128: one head, 64 query rows; wave w owns rows [w*16, w*16+16)
// K tile staged by TDM (tensor_load_to_lds), V tile staged transposed by VALU threads.
__global__ __launch_bounds__(128) void k_attn(const half_t* __restrict__ qh,
                                              const half_t* __restrict__ kh,
                                              const half_t* __restrict__ vh,
                                              const float* __restrict__ mask_loc,
                                              const float* __restrict__ mask_web,
                                              half_t* __restrict__ attn) {
  __shared__ __align__(16) half_t sK[32 * 64];     // [key][d], filled by TDM
  __shared__ __align__(16) half_t sVt[64 * 32];    // [d][key]
  __shared__ __align__(16) half_t sP[4][16 * 32];  // per-wave P tile [m][key]
  int tid = threadIdx.x, lane = tid & 31, wave = tid >> 5;
  int head = blockIdx.y, qb = blockIdx.x * 64;
  int kv = head >> 2;  // GROUPS = 4
  const float* mask = (head < 16) ? mask_loc : mask_web;
  int q0 = qb + wave * 16;
  int ln16 = lane & 15, mo = (lane >> 4) * 8;

  const half_t* qbase = qh + ((size_t)head * S_LEN + q0) * HD_;
  v16h Aq0 = load_frag_a(qbase, HD_, lane);       // d 0..31
  v16h Aq1 = load_frag_a(qbase + 32, HD_, lane);  // d 32..63

  v8f acc[4];
  float mrow[8], lrow[8];
  for (int t = 0; t < 4; ++t)
    for (int r = 0; r < 8; ++r) acc[t][r] = 0.f;
  for (int r = 0; r < 8; ++r) { mrow[r] = -1e30f; lrow[r] = 0.f; }
  const float scale = 0.125f;  // 1/sqrt(64)
  unsigned sK_lds = (unsigned)(uintptr_t)&sK[0];

  for (int kb = 0; kb < S_LEN; kb += 32) {
    // --- K tile via Tensor Data Mover (one wave issues; TENSORcnt + barrier publishes)
    if (wave == 0) {
      tdm_load_tile_f16(sK_lds, kh + ((size_t)kv * S_LEN + kb) * HD_, 32u, 64u, 64u);
      wait_tensorcnt0();
    }
    // --- V tile staged transposed by all threads
    int kr = tid >> 2, c = (tid & 3) * 16;
    const half_t* vg = vh + ((size_t)kv * S_LEN + kb + kr) * HD_ + c;
    v8h v0 = *(const v8h*)vg, v1 = *(const v8h*)(vg + 8);
#pragma unroll
    for (int j = 0; j < 8; ++j) {
      sVt[(c + j) * 32 + kr]     = v0[j];
      sVt[(c + 8 + j) * 32 + kr] = v1[j];
    }
    // prefetch next K/V tiles into L2 while we compute
    if (kb + 32 < S_LEN) {
      prefetch_ro(kh + ((size_t)kv * S_LEN + kb + 32) * HD_);
      prefetch_ro(vh + ((size_t)kv * S_LEN + kb + 32) * HD_);
    }
    __syncthreads();

    // scores: two 16x16 fragments (keys 0-15 and 16-31), contraction over d=64
    v8f s0, s1;
    for (int r = 0; r < 8; ++r) { s0[r] = 0.f; s1[r] = 0.f; }
    v16h b;
    b = load_frag_b(sK + 0, HD_, lane);             s0 = wmma_f16(Aq0, b, s0);
    b = load_frag_b(sK + 32, HD_, lane);            s0 = wmma_f16(Aq1, b, s0);
    b = load_frag_b(sK + 16 * HD_ + 0, HD_, lane);  s1 = wmma_f16(Aq0, b, s1);
    b = load_frag_b(sK + 16 * HD_ + 32, HD_, lane); s1 = wmma_f16(Aq1, b, s1);

    // streaming softmax on C-fragment layout (row M = r+mo, col N = ln16)
    half_t* P = sP[wave];
#pragma unroll
    for (int r = 0; r < 8; ++r) {
      int qrow = q0 + mo + r;
      float x0 = s0[r] * scale + mask[(size_t)qrow * S_LEN + kb + ln16];
      float x1 = s1[r] * scale + mask[(size_t)qrow * S_LEN + kb + 16 + ln16];
      float t = fmaxf(x0, x1);
#pragma unroll
      for (int o = 1; o < 16; o <<= 1) t = fmaxf(t, __shfl_xor(t, o, 16));
      float mn = fmaxf(mrow[r], t);
      float al = __expf(mrow[r] - mn);
      float p0 = __expf(x0 - mn), p1 = __expf(x1 - mn);
      float ps = p0 + p1;
#pragma unroll
      for (int o = 1; o < 16; o <<= 1) ps += __shfl_xor(ps, o, 16);
      lrow[r] = lrow[r] * al + ps;
      mrow[r] = mn;
      acc[0][r] *= al; acc[1][r] *= al; acc[2][r] *= al; acc[3][r] *= al;
      P[(mo + r) * 32 + ln16]      = (half_t)p0;
      P[(mo + r) * 32 + 16 + ln16] = (half_t)p1;
    }

    // out += P @ V  (P 16x32 as A-fragment, V-tiles as B-fragments)
    v16h Pa = load_frag_a(P, 32, lane);
#pragma unroll
    for (int dt = 0; dt < 4; ++dt) {
      v16h vb = load_frag_b(sVt + dt * 16 * 32, 32, lane);
      acc[dt] = wmma_f16(Pa, vb, acc[dt]);
    }
    __syncthreads();
  }

  // normalize; write [s, head*64+d] f16 for O-proj GEMM
#pragma unroll
  for (int dt = 0; dt < 4; ++dt)
#pragma unroll
    for (int r = 0; r < 8; ++r) {
      int row = q0 + mo + r;
      attn[(size_t)row * HIDN + head * HD_ + dt * 16 + ln16] = (half_t)(acc[dt][r] / lrow[r]);
    }
}

// ---------------------------------------------------------------- launch
extern "C" void kernel_launch(void* const* d_in, const int* in_sizes, int n_in,
                              void* d_out, int out_size, void* d_ws, size_t ws_size,
                              hipStream_t stream) {
  (void)in_sizes; (void)n_in; (void)out_size; (void)ws_size;
  const float* hidden   = (const float*)d_in[0];
  // d_in[1] = position_ids (int64) == arange(S); folded into the RoPE kernel
  const float* mask_loc = (const float*)d_in[2];
  const float* mask_web = (const float*)d_in[3];
  const float* Wq   = (const float*)d_in[4];
  const float* Wk   = (const float*)d_in[5];
  const float* Wv   = (const float*)d_in[6];
  const float* Wo   = (const float*)d_in[7];
  const float* qlnw = (const float*)d_in[8];
  const float* klnw = (const float*)d_in[9];

  char* w = (char*)d_ws;
  size_t off = 0;
  auto take = [&](size_t bytes) -> char* {
    char* p = w + off;
    off += (bytes + 255) & ~(size_t)255;
    return p;
  };
  half_t* xh  = (half_t*)take(sizeof(half_t) * S_LEN * HIDN);
  half_t* wqh = (half_t*)take(sizeof(half_t) * HIDN * NH * HD_);
  half_t* wkh = (half_t*)take(sizeof(half_t) * HIDN * NKV * HD_);
  half_t* wvh = (half_t*)take(sizeof(half_t) * HIDN * NKV * HD_);
  half_t* woh = (half_t*)take(sizeof(half_t) * NH * HD_ * HIDN);
  float*  qf  = (float*)take(sizeof(float) * S_LEN * NH * HD_);
  float*  kf  = (float*)take(sizeof(float) * S_LEN * NKV * HD_);
  float*  vf  = (float*)take(sizeof(float) * S_LEN * NKV * HD_);
  half_t* qhd = (half_t*)take(sizeof(half_t) * NH * S_LEN * HD_);
  half_t* khd = (half_t*)take(sizeof(half_t) * NKV * S_LEN * HD_);
  half_t* vhd = (half_t*)take(sizeof(half_t) * NKV * S_LEN * HD_);
  half_t* ath = (half_t*)take(sizeof(half_t) * S_LEN * HIDN);

  // 1) fp32 -> f16
  k_f32_to_f16<<<2048, 256, 0, stream>>>(hidden, xh, S_LEN * HIDN);
  k_f32_to_f16<<<2048, 256, 0, stream>>>(Wq, wqh, HIDN * NH * HD_);
  k_f32_to_f16<<<1024, 256, 0, stream>>>(Wk, wkh, HIDN * NKV * HD_);
  k_f32_to_f16<<<1024, 256, 0, stream>>>(Wv, wvh, HIDN * NKV * HD_);
  k_f32_to_f16<<<2048, 256, 0, stream>>>(Wo, woh, NH * HD_ * HIDN);

  // 2) QKV projections (WMMA GEMM)
  k_gemm<<<dim3(NH * HD_ / 64, S_LEN / 64), 128, 0, stream>>>(xh, wqh, qf, S_LEN, NH * HD_, HIDN);
  k_gemm<<<dim3(NKV * HD_ / 64, S_LEN / 64), 128, 0, stream>>>(xh, wkh, kf, S_LEN, NKV * HD_, HIDN);
  k_gemm<<<dim3(NKV * HD_ / 64, S_LEN / 64), 128, 0, stream>>>(xh, wvh, vf, S_LEN, NKV * HD_, HIDN);

  // 3) RoPE + RMSNorm, head-major f16
  k_rope_rms<<<dim3(S_LEN, NH), 64, 0, stream>>>(qf, qhd, qlnw, NH);
  k_rope_rms<<<dim3(S_LEN, NKV), 64, 0, stream>>>(kf, khd, klnw, NKV);
  k_v_pack<<<(S_LEN * NKV * HD_) / 256, 256, 0, stream>>>(vf, vhd);

  // 4) flash attention (TDM-staged K tiles, WMMA QK^T and PV, streaming softmax)
  k_attn<<<dim3(S_LEN / 64, NH), 128, 0, stream>>>(qhd, khd, vhd, mask_loc, mask_web, ath);

  // 5) output projection -> fp32 d_out
  k_gemm<<<dim3(HIDN / 64, S_LEN / 64), 128, 0, stream>>>(ath, woh, (float*)d_out, S_LEN, HIDN, HIDN);
}
